// DetectionHead_34256659153103
// MI455X (gfx1250) — matrix operand order repaired and checked by reading, not compiled, not run on hardware
//
#include <hip/hip_runtime.h>
#include <hip/hip_bf16.h>
#include <stdint.h>

// ---------------------------------------------------------------------------
// Detection head for MI455X (gfx1250, wave32, WMMA).
//   pooled = RoIAlign(features, proposals)         [1000, 256, 7, 7]
//   x = pooled.flatten (bf16, M padded to 1024)    [1024, 12544]
//   h1 = relu(x @ W1  + b1)    WMMA bf16 GEMM      [1024, 1024]
//   c1 = relu(x @ Wc1 + bc1)   WMMA bf16 GEMM      [1024, 1024]
//   h2 = relu(h1 @ W2 + b2)    WMMA bf16 GEMM      [1024, 1024]
//   logits = c1 @ Wc2 + bc2    WMMA, f32 out       [1000, 81]
//   bbox   = h2 @ W3  + b3     tiny f32 kernel     [1000, 4]
// Weights repacked fp32 [K][N] -> bf16 [Npad][K] so each WMMA fragment is
// two global_load_b128 per lane. Working set (~140MB) fits the 192MB L2.
// GEMM waves use a 32x64 tile (8 WMMA / 32-K step), double-buffered, with
// uniform-base + 32-bit per-lane offsets so loads use the SADDR form and
// the k-advance is a single scalar add (no 64-bit VALU address math).
// ---------------------------------------------------------------------------

#define IN_CH 256
#define FH 50
#define FW 50
#define KROI 1000
#define MPAD 1024
#define D_FEAT 12544        // 256*7*7
#define HID 1024
#define NCLS 81
#define NCLS_PAD 128

typedef __attribute__((ext_vector_type(16))) __bf16 v16bf;
typedef __attribute__((ext_vector_type(8)))  float  v8f;

union Frag16 {
    uint4 u[2];
    v16bf v;
};

static __device__ __forceinline__ uint16_t f2bf(float f) {
    // round-to-nearest-even f32 -> bf16
    uint32_t u = __float_as_uint(f);
    u += 0x7FFFu + ((u >> 16) & 1u);
    return (uint16_t)(u >> 16);
}
static __device__ __forceinline__ float bf2f(uint16_t h) {
    return __uint_as_float(((uint32_t)h) << 16);
}

// ---------------------------------------------------------------------------
// 0) NCHW -> NHWC transpose of features (fp32, 5MB total; L2-resident after).
// ---------------------------------------------------------------------------
__global__ void nchw_to_nhwc_kernel(const float* __restrict__ in,
                                    float* __restrict__ out) {
    int idx = blockIdx.x * blockDim.x + threadIdx.x;       // over N*H*W*C
    const int total = 2 * FH * FW * IN_CH;
    if (idx >= total) return;
    int c  = idx & (IN_CH - 1);
    int hw = (idx >> 8) % (FH * FW);
    int n  = idx / (IN_CH * FH * FW);
    out[idx] = in[((size_t)n * IN_CH + c) * (FH * FW) + hw];
}

// ---------------------------------------------------------------------------
// 1) Weight repack: W fp32 [K][N] row-major -> Wt bf16 [Npad][K] row-major.
//    (so each WMMA lane reads K-contiguous 16B chunks). Zero-fills n >= N.
// ---------------------------------------------------------------------------
__global__ void pack_w_kernel(const float* __restrict__ W,
                              uint16_t* __restrict__ Wt,
                              int K, int N) {
    __shared__ float tile[32][33];
    const int tx = threadIdx.x & 31;
    const int ty = threadIdx.x >> 5;          // 0..7
    const int k0 = blockIdx.x * 32;
    const int n0 = blockIdx.y * 32;
#pragma unroll
    for (int i = 0; i < 4; ++i) {
        int k = k0 + ty + i * 8;
        int n = n0 + tx;
        tile[ty + i * 8][tx] = (n < N) ? W[(size_t)k * N + n] : 0.0f;
    }
    __syncthreads();
#pragma unroll
    for (int i = 0; i < 4; ++i) {
        int n = n0 + ty + i * 8;              // row of Wt (< Npad by grid)
        int k = k0 + tx;
        Wt[(size_t)n * K + k] = f2bf(tile[tx][ty + i * 8]);
    }
}

// ---------------------------------------------------------------------------
// 2) RoIAlign (torchvision, aligned=false, SR=2, scale=1/16) -> bf16 rows.
//    One block per (padded) ROI; 256 threads = 256 channels. Row staged in
//    LDS then stored with coalesced b128 writes. Rows >= KROI are zeroed.
// ---------------------------------------------------------------------------
__global__ void roi_align_kernel(const float* __restrict__ feat_nhwc,
                                 const float* __restrict__ props,
                                 uint16_t* __restrict__ xout) {
    const int roi = blockIdx.x;
    const int tid = threadIdx.x;              // channel
    __shared__ __align__(16) uint16_t srow[D_FEAT];
    uint16_t* outrow = xout + (size_t)roi * D_FEAT;

    if (roi >= KROI) {
        uint4 z = make_uint4(0u, 0u, 0u, 0u);
        for (int i = tid; i < D_FEAT / 8; i += 256)
            ((uint4*)outrow)[i] = z;
        return;
    }

    const float* p = props + roi * 5;
    const int   b  = (int)p[0];
    const float x1 = p[1] * 0.0625f;
    const float y1 = p[2] * 0.0625f;
    const float x2 = p[3] * 0.0625f;
    const float y2 = p[4] * 0.0625f;
    const float rw = fmaxf(x2 - x1, 1.0f);
    const float rh = fmaxf(y2 - y1, 1.0f);
    const float bw = rw * (1.0f / 7.0f);
    const float bh = rh * (1.0f / 7.0f);
    const float* fb = feat_nhwc + (size_t)b * FH * FW * IN_CH;
    const int c = tid;

    for (int ph = 0; ph < 7; ++ph) {
        for (int pw = 0; pw < 7; ++pw) {
            float acc = 0.0f;
#pragma unroll
            for (int sy = 0; sy < 2; ++sy) {
#pragma unroll
                for (int sx = 0; sx < 2; ++sx) {
                    float yy = y1 + ((float)ph + (sy + 0.5f) * 0.5f) * bh;
                    float xx = x1 + ((float)pw + (sx + 0.5f) * 0.5f) * bw;
                    bool v = (yy > -1.0f) && (yy < (float)FH) &&
                             (xx > -1.0f) && (xx < (float)FW);
                    if (v) {
                        float yc = fminf(fmaxf(yy, 0.0f), (float)(FH - 1));
                        float xc = fminf(fmaxf(xx, 0.0f), (float)(FW - 1));
                        int yi0 = (int)floorf(yc);
                        int xi0 = (int)floorf(xc);
                        int yi1 = min(yi0 + 1, FH - 1);
                        int xi1 = min(xi0 + 1, FW - 1);
                        float ly = yc - (float)yi0, lx = xc - (float)xi0;
                        float hy = 1.0f - ly,       hx = 1.0f - lx;
                        float f00 = fb[((size_t)yi0 * FW + xi0) * IN_CH + c];
                        float f01 = fb[((size_t)yi0 * FW + xi1) * IN_CH + c];
                        float f10 = fb[((size_t)yi1 * FW + xi0) * IN_CH + c];
                        float f11 = fb[((size_t)yi1 * FW + xi1) * IN_CH + c];
                        acc += hy * hx * f00 + hy * lx * f01 +
                               ly * hx * f10 + ly * lx * f11;
                    }
                }
            }
            srow[c * 49 + ph * 7 + pw] = f2bf(acc * 0.25f);
        }
    }
    __syncthreads();
    for (int i = tid; i < D_FEAT / 8; i += 256)
        ((uint4*)outrow)[i] = ((const uint4*)srow)[i];
}

// ---------------------------------------------------------------------------
// 3) WMMA GEMM:  OUT[M x N] = act(A[M x K] @ Bt^T + bias)
//    A  : bf16 [MPAD][K] row-major
//    Bt : bf16 [Npad][K] row-major (pre-packed transposed weights)
//    256 threads = 8 waves; block tile 256(M) x 64(N); each wave: 32x64 =
//    8 x v_wmma_f32_16x16x32_bf16 per 32-wide K step, double-buffered.
//    Addressing: uniform base (SGPR) + 32-bit per-lane byte offsets so the
//    k-advance folds into the scalar base instead of 64-bit VALU adds.
//    Requires K % 64 == 0 and K >= 64 (true here: 12544, 1024).
// ---------------------------------------------------------------------------
#define LOAD_SET(aF, bF, kk)                                                 \
    do {                                                                     \
        _Pragma("unroll")                                                    \
        for (int tm = 0; tm < 2; ++tm) {                                     \
            aF[tm].u[0] = *(const uint4*)(Ab + (aoff[tm] + 2 * (kk)));       \
            aF[tm].u[1] = *(const uint4*)(Ab + (aoff[tm] + 2 * (kk) + 32));  \
        }                                                                    \
        _Pragma("unroll")                                                    \
        for (int t = 0; t < 4; ++t) {                                        \
            bF[t].u[0] = *(const uint4*)(Bb + (boff[t] + 2 * (kk)));         \
            bF[t].u[1] = *(const uint4*)(Bb + (boff[t] + 2 * (kk) + 32));    \
        }                                                                    \
    } while (0)

#define MMA_SET(aF, bF)                                                      \
    do {                                                                     \
        _Pragma("unroll")                                                    \
        for (int tm = 0; tm < 2; ++tm) {                                     \
            _Pragma("unroll")                                                \
            for (int t = 0; t < 4; ++t)                                      \
                acc[tm][t] = __builtin_amdgcn_wmma_f32_16x16x32_bf16(        \
                    false, aF[tm].v, false, bF[t].v, (short)0,               \
                    acc[tm][t], false, false);                               \
        }                                                                    \
    } while (0)

template <bool RELU, bool BF16OUT>
__global__ __launch_bounds__(256)
void gemm_wmma_kernel(const uint16_t* __restrict__ A,
                      const uint16_t* __restrict__ Bt,
                      const float* __restrict__ bias,
                      void* __restrict__ outp,
                      int K, int ldo, int Mreal, int Nreal) {
    const int lane = threadIdx.x & 31;
    const int wave = threadIdx.x >> 5;                 // 0..7
    const int m0 = blockIdx.y * 256 + wave * 32;
    const int n0 = blockIdx.x * 64;
    const int lh = lane >> 4;                          // 0/1 (lane half)
    const int ll = lane & 15;

    const char* Ab = (const char*)A;                   // uniform SGPR base
    const char* Bb = (const char*)Bt;                  // uniform SGPR base

    // A fragment (16x32 bf16): lanes 0-15 row m, v0-3=K0..7, v4-7=K16..23;
    // lanes 16-31 same rows, K8..15 / K24..31.  (byte offsets, fit in i32)
    int aoff[2];
    aoff[0] = ((m0 + ll) * K + lh * 8) * 2;
    aoff[1] = ((m0 + 16 + ll) * K + lh * 8) * 2;
    // B fragment (32x16 bf16): lanes 0-15 col n K0..7, K16..23; lanes 16-31 K8..15, K24..31.
    int boff[4];
#pragma unroll
    for (int t = 0; t < 4; ++t)
        boff[t] = ((n0 + t * 16 + ll) * K + lh * 8) * 2;

    v8f acc[2][4];
#pragma unroll
    for (int tm = 0; tm < 2; ++tm)
#pragma unroll
        for (int t = 0; t < 4; ++t)
            acc[tm][t] = (v8f){0.f, 0.f, 0.f, 0.f, 0.f, 0.f, 0.f, 0.f};

    Frag16 a0[2], b0[4], a1[2], b1[4];

    // ping-pong pipeline: set0 always holds k0's data at loop top
    LOAD_SET(a0, b0, 0);
    int k0 = 0;
    for (; k0 < K - 64; k0 += 64) {
        LOAD_SET(a1, b1, k0 + 32);
        MMA_SET(a0, b0);
        LOAD_SET(a0, b0, k0 + 64);
        MMA_SET(a1, b1);
    }
    LOAD_SET(a1, b1, k0 + 32);
    MMA_SET(a0, b0);
    MMA_SET(a1, b1);

    // C/D layout: lane l, vgpr r -> (M = r + 8*lh, N = ll)
#pragma unroll
    for (int tm = 0; tm < 2; ++tm) {
#pragma unroll
        for (int t = 0; t < 4; ++t) {
            int col = n0 + t * 16 + ll;
            float bv = (col < Nreal) ? bias[col] : 0.0f;
#pragma unroll
            for (int r = 0; r < 8; ++r) {
                int row = m0 + tm * 16 + lh * 8 + r;
                float v = acc[tm][t][r] + bv;
                if (RELU) v = fmaxf(v, 0.0f);
                if (BF16OUT) {
                    ((uint16_t*)outp)[(size_t)row * ldo + col] = f2bf(v);
                } else {
                    if (row < Mreal && col < Nreal)
                        ((float*)outp)[(size_t)row * ldo + col] = v;
                }
            }
        }
    }
}

// ---------------------------------------------------------------------------
// 4) bbox head: out[1000,4] = h2(bf16)[1000,1024] @ W3[1024,4] + b3
// ---------------------------------------------------------------------------
__global__ void bbox_kernel(const uint16_t* __restrict__ h2,
                            const float* __restrict__ W3,
                            const float* __restrict__ b3,
                            float* __restrict__ out) {
    int idx = blockIdx.x * blockDim.x + threadIdx.x;
    if (idx >= KROI * 4) return;
    int row = idx >> 2;
    int j = idx & 3;
    const uint16_t* hr = h2 + (size_t)row * HID;
    float s = b3[j];
    for (int k = 0; k < HID; ++k)
        s += bf2f(hr[k]) * W3[k * 4 + j];
    out[idx] = s;
}

// ---------------------------------------------------------------------------
// launcher
// ---------------------------------------------------------------------------
extern "C" void kernel_launch(void* const* d_in, const int* in_sizes, int n_in,
                              void* d_out, int out_size, void* d_ws, size_t ws_size,
                              hipStream_t stream) {
    const float* features  = (const float*)d_in[0];   // [2,256,50,50]
    const float* proposals = (const float*)d_in[1];   // [1000,5]
    const float* W1  = (const float*)d_in[2];         // [12544,1024]
    const float* b1  = (const float*)d_in[3];
    const float* W2  = (const float*)d_in[4];         // [1024,1024]
    const float* b2  = (const float*)d_in[5];
    const float* W3  = (const float*)d_in[6];         // [1024,4]
    const float* b3  = (const float*)d_in[7];
    const float* Wc1 = (const float*)d_in[8];         // [12544,1024]
    const float* bc1 = (const float*)d_in[9];
    const float* Wc2 = (const float*)d_in[10];        // [1024,81]
    const float* bc2 = (const float*)d_in[11];

    float* out_bbox   = (float*)d_out;                // [1000*4]
    float* out_logits = (float*)d_out + KROI * 4;     // [1000*81]

    // workspace carve-out (256B aligned)
    char* ws = (char*)d_ws;
    size_t off = 0;
    auto take = [&](size_t bytes) -> char* {
        char* r = ws + off;
        off += (bytes + 255) & ~(size_t)255;
        return r;
    };
    float*    feat_nhwc = (float*)   take((size_t)2 * FH * FW * IN_CH * 4);
    uint16_t* xbf       = (uint16_t*)take((size_t)MPAD * D_FEAT * 2);
    uint16_t* W1t       = (uint16_t*)take((size_t)HID * D_FEAT * 2);
    uint16_t* Wc1t      = (uint16_t*)take((size_t)HID * D_FEAT * 2);
    uint16_t* W2t       = (uint16_t*)take((size_t)HID * HID * 2);
    uint16_t* Wc2t      = (uint16_t*)take((size_t)NCLS_PAD * HID * 2);
    uint16_t* h1        = (uint16_t*)take((size_t)MPAD * HID * 2);
    uint16_t* c1        = (uint16_t*)take((size_t)MPAD * HID * 2);
    uint16_t* h2        = (uint16_t*)take((size_t)MPAD * HID * 2);
    (void)ws_size; (void)in_sizes; (void)n_in; (void)out_size;

    // 0) feature layout transform
    {
        int total = 2 * FH * FW * IN_CH;
        nchw_to_nhwc_kernel<<<(total + 255) / 256, 256, 0, stream>>>(features, feat_nhwc);
    }
    // 1) weight repacks (fp32 [K][N] -> bf16 [Npad][K])
    pack_w_kernel<<<dim3(D_FEAT / 32, HID / 32), 256, 0, stream>>>(W1,  W1t,  D_FEAT, HID);
    pack_w_kernel<<<dim3(D_FEAT / 32, HID / 32), 256, 0, stream>>>(Wc1, Wc1t, D_FEAT, HID);
    pack_w_kernel<<<dim3(HID / 32,    HID / 32), 256, 0, stream>>>(W2,  W2t,  HID, HID);
    pack_w_kernel<<<dim3(HID / 32, NCLS_PAD / 32), 256, 0, stream>>>(Wc2, Wc2t, HID, NCLS);
    // 2) RoIAlign -> bf16 rows (padded to 1024 rows)
    roi_align_kernel<<<MPAD, 256, 0, stream>>>(feat_nhwc, proposals, xbf);
    // 3) WMMA GEMMs (block tile 256x64, wave tile 32x64)
    gemm_wmma_kernel<true, true><<<dim3(HID / 64, MPAD / 256), 256, 0, stream>>>(
        xbf, W1t, b1, h1, D_FEAT, HID, MPAD, HID);
    gemm_wmma_kernel<true, true><<<dim3(HID / 64, MPAD / 256), 256, 0, stream>>>(
        xbf, Wc1t, bc1, c1, D_FEAT, HID, MPAD, HID);
    gemm_wmma_kernel<true, true><<<dim3(HID / 64, MPAD / 256), 256, 0, stream>>>(
        h1, W2t, b2, h2, HID, HID, MPAD, HID);
    gemm_wmma_kernel<false, false><<<dim3(NCLS_PAD / 64, MPAD / 256), 256, 0, stream>>>(
        c1, Wc2t, bc2, out_logits, HID, NCLS, KROI, NCLS);
    // 4) bbox head
    bbox_kernel<<<(KROI * 4 + 255) / 256, 256, 0, stream>>>(h2, W3, b3, out_bbox);
}
